// RNN_Memory_57509612093969
// MI455X (gfx1250) — matrix operand order repaired and checked by reading, not compiled, not run on hardware
//
#include <hip/hip_runtime.h>

// ---------------- types ----------------
typedef __bf16 bf16;
typedef __attribute__((ext_vector_type(16))) __bf16 v16bf;
typedef __attribute__((ext_vector_type(8)))  __bf16 v8bf;
typedef __attribute__((ext_vector_type(8)))  float  v8f;
typedef __attribute__((ext_vector_type(4)))  float  v4f;
typedef __attribute__((ext_vector_type(4)))  unsigned int v4u;
typedef __attribute__((ext_vector_type(4)))  int    v4i;
typedef __attribute__((ext_vector_type(8)))  int    v8i;

#if defined(__has_builtin)
#if __has_builtin(__builtin_amdgcn_tensor_load_to_lds)
#define HAVE_TDM 1
#else
#define HAVE_TDM 0
#endif
#else
#define HAVE_TDM 0
#endif

// ---------------- problem dims ----------------
constexpr int T   = 128;
constexpr int Bsz = 64;
constexpr int Isz = 256;
constexpr int Hsz = 512;
constexpr int Osz = 1000;
constexpr int G4  = 4 * Hsz;          // 2048 gate columns

constexpr int SLICE = 16;             // hidden units per gate-block
constexpr int NSLC  = Hsz / SLICE;    // 32 hidden slices
constexpr int KC    = 128;            // K chunk per gate-block
constexpr int NKC   = 6;              // 2 chunks from x/W_ih + 4 from h/W_hh
// grid per step = NSLC x NKC = 192 blocks -> enough WGPs in flight to pull
// the 6.3 MB/step weight stream at HBM rate instead of bottlenecking on 8 WGPs.

// LDS strides (elements), padded; byte strides stay 16B-aligned
constexpr int ASTR = KC + 8;          // 136 (bf16) -> 272B row stride
constexpr int WSTR = KC + 8;          // 136 (bf16)

// projection
constexpr int NPAD  = 1008;           // 63 * 16, O padded
constexpr int LSTR  = NPAD + 8;       // 1016 f32
constexpr int PASTR = Hsz + 8;        // 520 bf16

__device__ __forceinline__ v16bf cat8(v8bf lo, v8bf hi) {
  return __builtin_shufflevector(lo, hi, 0,1,2,3,4,5,6,7,8,9,10,11,12,13,14,15);
}

// ---------------- TDM: 2D bf16 tile -> LDS with row padding ----------------
// D# per CDNA5 ISA 8.3/8.4: group0 {count=1, lds_addr, global_addr, type=2},
// group1 {data_size=1(2B), pad_enable, pad codes, tensor dims, tile dims, stride0}.
__device__ __forceinline__ void tdm_load_2d_bf16(const void* lds_dst, const void* gsrc,
                                                 unsigned tensor_d0, unsigned tensor_d1,
                                                 unsigned tile_d0, unsigned tile_d1,
                                                 unsigned stride0,
                                                 unsigned pad_interval_code,
                                                 unsigned pad_amount_code)
{
#if HAVE_TDM
  const unsigned long long ga = (unsigned long long)gsrc;
  const unsigned lds_off = (unsigned)(unsigned long long)lds_dst; // LDS aperture: addr[31:0]
  v4u g0;
  g0.x = 1u;                                                   // count=1 (valid D#)
  g0.y = lds_off;                                              // lds_addr (bytes)
  g0.z = (unsigned)ga;                                         // global_addr[31:0]
  g0.w = ((unsigned)(ga >> 32) & 0x01FFFFFFu) | (2u << 30);    // global_addr[56:32] | type=2
  v8i g1;
  g1[0] = (int)((1u << 16)                      // data_size = 1 -> 2 bytes (bf16)
              | (1u << 20)                      // pad_enable
              | (pad_interval_code << 22)
              | (pad_amount_code << 25));
  g1[1] = (int)((tensor_d0 & 0xFFFFu) << 16);
  g1[2] = (int)(((tensor_d0 >> 16) & 0xFFFFu) | ((tensor_d1 & 0xFFFFu) << 16));
  g1[3] = (int)(((tensor_d1 >> 16) & 0xFFFFu) | ((tile_d0 & 0xFFFFu) << 16));
  g1[4] = (int)(tile_d1 & 0xFFFFu);
  g1[5] = (int)stride0;
  g1[6] = 0;
  g1[7] = 0;
  const v4i gz = {0, 0, 0, 0};
#if __clang_major__ >= 23
  const v8i gz8 = {0, 0, 0, 0, 0, 0, 0, 0};
  __builtin_amdgcn_tensor_load_to_lds(g0, g1, gz, gz, gz8, 0);
#else
  __builtin_amdgcn_tensor_load_to_lds(g0, g1, gz, gz, 0);
#endif
#endif
}

// ---------------- init: h0/c0 seed + Wo -> bf16 (padded to 1008 rows) ----------------
__global__ __launch_bounds__(256)
void init_kernel(const float* __restrict__ Wo, const float* __restrict__ h0,
                 const float* __restrict__ c0,
                 bf16* __restrict__ Wo_bf, bf16* __restrict__ hp0, bf16* __restrict__ hp1,
                 float* __restrict__ hout, float* __restrict__ cout)
{
  const int stride = gridDim.x * blockDim.x;
  for (int i = blockIdx.x * blockDim.x + threadIdx.x; i < NPAD * Hsz; i += stride) {
    const int r = i >> 9;   // / Hsz
    Wo_bf[i] = (r < Osz) ? (bf16)Wo[i] : (bf16)0.0f;
    if (i < Bsz * Hsz) {
      const float h = h0[i], c = c0[i];
      hout[i] = h; cout[i] = c;
      hp0[i] = (bf16)h; hp1[i] = (bf16)h;
    }
  }
}

// ---------------- gate partial GEMM (per timestep) ----------------
// grid = (NSLC=32, NKC=6) x 256 threads (8 wave32).
// Block (slice, kblk): partial gates for hidden cols [n0,n0+16) of all 4 gates over one
// 128-wide K chunk: pg[kblk][B=64][gate cols]. A tile comes from x (f32->bf16) or
// h_prev (TDM bf16 copy); W tile is streamed f32->bf16 once (memory-bound roofline path).
__global__ __launch_bounds__(256)
void lstm_gate_kernel(const float* __restrict__ xt,    // [B, I] f32
                      const float* __restrict__ Wih,   // [4H, I] f32
                      const float* __restrict__ Whh,   // [4H, H] f32
                      const bf16*  __restrict__ hprev, // [B, H] bf16
                      float*       __restrict__ pg)    // [NKC][B][4H] f32 partials
{
  extern __shared__ char smem[];
  bf16* As = (bf16*)smem;                              // [64][ASTR]
  bf16* Ws = (bf16*)(smem + 64 * ASTR * sizeof(bf16)); // [64][WSTR] (4 gates x 16 rows)

  const int tid   = threadIdx.x;
  const int slice = blockIdx.x;
  const int kblk  = blockIdx.y;
  const int n0    = slice * SLICE;

  // ---- stage A chunk [64 x 128] bf16
  if (kblk < 2) {
    for (int idx = tid; idx < 64 * 32; idx += 256) {
      const int r = idx >> 5, k4 = (idx & 31) << 2;
      const v4f v = *(const v4f*)(xt + r * Isz + kblk * KC + k4);
      bf16* dst = As + r * ASTR + k4;
      dst[0] = (bf16)v.x; dst[1] = (bf16)v.y; dst[2] = (bf16)v.z; dst[3] = (bf16)v.w;
    }
  } else {
#if HAVE_TDM
    // pure bf16 tile copy -> Tensor Data Mover, issued once by wave 0.
    // LDS row padding by TDM: interval 64 dwords (code 5), amount 4 dwords (code 3).
    if (tid < 32) {
      tdm_load_2d_bf16(As, hprev + (kblk - 2) * KC,
                       /*tensor_d0=*/Hsz, /*tensor_d1=*/Bsz,
                       /*tile_d0=*/KC, /*tile_d1=*/Bsz,
                       /*stride0=*/Hsz,
                       /*pad_interval=*/5u, /*pad_amount=*/3u);
    }
#else
    for (int idx = tid; idx < 64 * 16; idx += 256) {
      const int r = idx >> 4, k8 = (idx & 15) << 3;
      *(v8bf*)(As + r * ASTR + k8) =
          *(const v8bf*)(hprev + r * Hsz + (kblk - 2) * KC + k8);
    }
#endif
  }

  // ---- stage W tile [64 gate rows x 128] f32 -> bf16 (overlaps the TDM copy)
  const float* wb; int ld; size_t tstride;
  if (kblk < 2) { wb = Wih + kblk * KC;       ld = Isz; tstride = (size_t)G4 * Isz; }
  else          { wb = Whh + (kblk - 2) * KC; ld = Hsz; tstride = (size_t)G4 * Hsz; }
  for (int idx = tid; idx < 64 * 32; idx += 256) {
    const int r = idx >> 5, k4 = (idx & 31) << 2;
    const int g = r >> 4, i = r & 15;
    const int R = g * Hsz + n0 + i;
    const v4f v = *(const v4f*)(wb + (size_t)R * ld + k4);
    bf16* dst = Ws + r * WSTR + k4;
    dst[0] = (bf16)v.x; dst[1] = (bf16)v.y; dst[2] = (bf16)v.z; dst[3] = (bf16)v.w;
  }
  // ---- prefetch next timestep's weight tile for this block into L2
  if (tid < 64) {
    const int g = tid >> 4, i = tid & 15;
    const int R = g * Hsz + n0 + i;
    const float* wn = wb + tstride + (size_t)R * ld;
    __builtin_prefetch(wn,      0, 1);
    __builtin_prefetch(wn + 64, 0, 1);
  }
#if HAVE_TDM
  if (kblk >= 2 && tid < 32) __builtin_amdgcn_s_wait_tensorcnt(0);
#endif
  __syncthreads();

  // ---- WMMA: 4 K-substeps of 32; each wave owns (mt, 2 gate tiles)
  const int lane = tid & 31;
  const int half = lane >> 4;
  const int l16  = lane & 15;
  const int wave = tid >> 5;
  const int mt   = wave & 3;            // 16-row M tile of B=64
  const int ntb  = (wave >> 2) * 2;     // gates {ntb, ntb+1}

  const v8f vzero = {0.f,0.f,0.f,0.f,0.f,0.f,0.f,0.f};
  v8f acc[2] = {vzero, vzero};
  #pragma unroll
  for (int ks = 0; ks < 4; ++ks) {
    const int k0 = ks * 32;
    const bf16* ap = As + (mt * 16 + l16) * ASTR + k0 + half * 16;
    const v16bf a = cat8(*(const v8bf*)ap, *(const v8bf*)(ap + 8));
    v16bf bfr[2];
    #pragma unroll
    for (int j = 0; j < 2; ++j) {
      const bf16* bp = Ws + ((ntb + j) * 16 + l16) * WSTR + k0 + half * 16;
      bfr[j] = cat8(*(const v8bf*)bp, *(const v8bf*)(bp + 8));
    }
    #pragma unroll
    for (int j = 0; j < 2; ++j) {
      acc[j] = __builtin_amdgcn_wmma_f32_16x16x32_bf16(
          false, a, false, bfr[j], (short)0, acc[j], false, false);
    }
  }

  // ---- write partial gates (C layout: VGPR v -> M = v + half*8, N = l16)
  float* pgk = pg + (size_t)kblk * Bsz * G4;
  #pragma unroll
  for (int j = 0; j < 2; ++j) {
    const int col = (ntb + j) * Hsz + n0 + l16;
    #pragma unroll
    for (int v = 0; v < 8; ++v) {
      const int m = mt * 16 + half * 8 + v;
      pgk[(size_t)m * G4 + col] = acc[j][v];
    }
  }
}

// ---------------- cell update (per timestep) ----------------
// grid = B*H/256 = 128 blocks. Sums the NKC partials in fixed order (deterministic),
// adds biases, applies the LSTM cell (PyTorch gate order i,f,g,o).
__global__ __launch_bounds__(256)
void lstm_cell_kernel(const float* __restrict__ pg,   // [NKC][B][4H]
                      const float* __restrict__ bih,  // [4H]
                      const float* __restrict__ bhh,  // [4H]
                      bf16*  __restrict__ hnext,      // [B,H] bf16
                      bf16*  __restrict__ hs_t,       // [B,H] bf16
                      float* __restrict__ hf32,       // [B,H] running h
                      float* __restrict__ cf32)       // [B,H] running c
{
  const int e  = blockIdx.x * 256 + threadIdx.x;   // 0..B*H-1
  const int b  = e >> 9;
  const int nh = e & (Hsz - 1);
  float gate[4];
  #pragma unroll
  for (int g = 0; g < 4; ++g) {
    const int col = g * Hsz + nh;
    float s = bih[col] + bhh[col];
    #pragma unroll
    for (int k = 0; k < NKC; ++k)
      s += pg[(size_t)k * Bsz * G4 + (size_t)b * G4 + col];
    gate[g] = s;
  }
  const float iv = 1.f / (1.f + __expf(-gate[0]));
  const float fv = 1.f / (1.f + __expf(-gate[1]));
  const float gv = tanhf(gate[2]);
  const float ov = 1.f / (1.f + __expf(-gate[3]));
  const float cN = fv * cf32[e] + iv * gv;
  const float hN = ov * tanhf(cN);
  cf32[e] = cN;
  hf32[e] = hN;
  const bf16 hb = (bf16)hN;
  hnext[e] = hb;
  hs_t[e]  = hb;
}

// ---------------- projection + log_softmax ----------------
// grid = T*B/16 = 512 blocks x 256 threads. Each block: 16 rows of hs (bf16) x Wo^T,
// logits staged in LDS, then row-wise log-softmax. Wo_bf (1MB) is L2-resident.
// hs tile staged via TDM (interval 256 dwords (code 7), pad 4 dwords (code 3)).
__global__ __launch_bounds__(256)
void proj_softmax_kernel(const bf16* __restrict__ hs,     // [T*B, H] bf16
                         const bf16* __restrict__ Wo_bf,  // [NPAD, H] bf16
                         const float* __restrict__ bo,    // [O]
                         float* __restrict__ out)         // [T*B, O]
{
  extern __shared__ char smem[];
  bf16*  As  = (bf16*)smem;                                   // [16][PASTR]
  float* Ls  = (float*)(smem + 16 * PASTR * sizeof(bf16));    // [16][LSTR]
  float* red = Ls + 16 * LSTR;                                // 256 + 16 + 16

  const int tid  = threadIdx.x;
  const int wave = tid >> 5;
  const int lane = tid & 31;
  const int half = lane >> 4;
  const int l16  = lane & 15;
  const int r0   = blockIdx.x * 16;

#if HAVE_TDM
  if (tid < 32) {
    tdm_load_2d_bf16(As, hs + (size_t)r0 * Hsz,
                     /*tensor_d0=*/Hsz, /*tensor_d1=*/(unsigned)(T * Bsz),
                     /*tile_d0=*/Hsz, /*tile_d1=*/16,
                     /*stride0=*/Hsz,
                     /*pad_interval=*/7u, /*pad_amount=*/3u);
    __builtin_amdgcn_s_wait_tensorcnt(0);
  }
#else
  for (int idx = tid; idx < 16 * 64; idx += 256) {
    const int r = idx >> 6, k8 = (idx & 63) << 3;
    *(v8bf*)(As + r * PASTR + k8) = *(const v8bf*)(hs + (size_t)(r0 + r) * Hsz + k8);
  }
#endif
  __syncthreads();

  const v8f vzero = {0.f,0.f,0.f,0.f,0.f,0.f,0.f,0.f};
  for (int nt = wave; nt < 63; nt += 8) {   // wave-uniform
    v8f acc = vzero;
    #pragma unroll 4
    for (int k0 = 0; k0 < Hsz; k0 += 32) {
      const bf16* ap = As + l16 * PASTR + k0 + half * 16;
      const v16bf a = cat8(*(const v8bf*)ap, *(const v8bf*)(ap + 8));
      const bf16* bp = Wo_bf + (size_t)(nt * 16 + l16) * Hsz + k0 + half * 16;
      const v16bf b = cat8(*(const v8bf*)bp, *(const v8bf*)(bp + 8));
      acc = __builtin_amdgcn_wmma_f32_16x16x32_bf16(
          false, a, false, b, (short)0, acc, false, false);
    }
    #pragma unroll
    for (int v = 0; v < 8; ++v) {
      const int m = half * 8 + v;
      const int c = nt * 16 + l16;
      float val = (c < Osz) ? (acc[v] + bo[c]) : -3.0e38f;
      Ls[m * LSTR + c] = val;
    }
  }
  __syncthreads();

  // row-wise log-softmax over 1000 valid logits; 16 threads per row
  const int row = tid >> 4, sub = tid & 15;
  float pm = -3.0e38f;
  for (int c = sub; c < Osz; c += 16) pm = fmaxf(pm, Ls[row * LSTR + c]);
  red[sub * 16 + row] = pm;
  __syncthreads();
  if (tid < 16) {
    float m = -3.0e38f;
    for (int s = 0; s < 16; ++s) m = fmaxf(m, red[s * 16 + tid]);
    red[256 + tid] = m;
  }
  __syncthreads();
  const float rm = red[256 + row];
  float ps = 0.f;
  for (int c = sub; c < Osz; c += 16) ps += __expf(Ls[row * LSTR + c] - rm);
  __syncthreads();
  red[sub * 16 + row] = ps;
  __syncthreads();
  if (tid < 16) {
    float s = 0.f;
    for (int q = 0; q < 16; ++q) s += red[q * 16 + tid];
    red[272 + tid] = __logf(s);
  }
  __syncthreads();
  const float lz = rm + red[272 + row];
  for (int c = sub; c < Osz; c += 16)
    out[(size_t)(r0 + row) * Osz + c] = Ls[row * LSTR + c] - lz;
}

// ---------------- host launch ----------------
extern "C" void kernel_launch(void* const* d_in, const int* in_sizes, int n_in,
                              void* d_out, int out_size, void* d_ws, size_t ws_size,
                              hipStream_t stream)
{
  const float* x   = (const float*)d_in[0];   // [T,B,I]
  const float* h0  = (const float*)d_in[1];   // [B,H]
  const float* c0  = (const float*)d_in[2];   // [B,H]
  const float* Wih = (const float*)d_in[3];   // [T,4H,I]
  const float* Whh = (const float*)d_in[4];   // [T,4H,H]
  const float* bih = (const float*)d_in[5];   // [T,4H]
  const float* bhh = (const float*)d_in[6];   // [T,4H]
  const float* Wo  = (const float*)d_in[7];   // [O,H]
  const float* bo  = (const float*)d_in[8];   // [O]
  // d_in[9] = input_lengths: not used by the reference outputs

  float* out   = (float*)d_out;
  float* o_out = out;                                   // [T,B,O]
  float* h_out = out + (size_t)T * Bsz * Osz;           // [B,H]
  float* c_out = h_out + (size_t)Bsz * Hsz;             // [B,H]

  // workspace: hs bf16 | h ping | h pong | Wo bf16 | gate partials f32
  char* ws = (char*)d_ws;
  bf16*  hs_bf = (bf16*)ws;
  bf16*  hp0   = (bf16*)(ws + (size_t)T * Bsz * Hsz * sizeof(bf16));
  bf16*  hp1   = hp0 + (size_t)Bsz * Hsz;
  bf16*  Wo_bf = hp1 + (size_t)Bsz * Hsz;
  float* pg    = (float*)(Wo_bf + (size_t)NPAD * Hsz);  // [NKC][B][4H]

  init_kernel<<<1024, 256, 0, stream>>>(Wo, h0, c0, Wo_bf, hp0, hp1, h_out, c_out);

  const size_t lds_gate = 64 * ASTR * sizeof(bf16) + 64 * WSTR * sizeof(bf16);
  for (int t = 0; t < T; ++t) {
    const bf16* hprev = (t & 1) ? hp1 : hp0;
    bf16*       hnext = (t & 1) ? hp0 : hp1;
    lstm_gate_kernel<<<dim3(NSLC, NKC), 256, lds_gate, stream>>>(
        x   + (size_t)t * Bsz * Isz,
        Wih + (size_t)t * G4 * Isz,
        Whh + (size_t)t * G4 * Hsz,
        hprev, pg);
    lstm_cell_kernel<<<(Bsz * Hsz) / 256, 256, 0, stream>>>(
        pg,
        bih + (size_t)t * G4,
        bhh + (size_t)t * G4,
        hnext,
        hs_bf + (size_t)t * Bsz * Hsz,
        h_out, c_out);
  }

  const size_t lds_proj = 16 * PASTR * sizeof(bf16)
                        + 16 * LSTR * sizeof(float)
                        + 288 * sizeof(float);
  proj_softmax_kernel<<<(T * Bsz) / 16, 256, lds_proj, stream>>>(hs_bf, Wo_bf, bo, o_out);
}
